// MultiHeadSelfAttention_26749056319610
// MI455X (gfx1250) — matrix operand order repaired and checked by reading, not compiled, not run on hardware
//
#include <hip/hip_runtime.h>

// MHSA: B=2, C=512, H=W=64 -> N=4096, 8 heads x 64 head_dim.
// f32->f16 staging, QKV GEMM (WMMA f16/f32acc), flash attention with
// async-to-LDS double-buffered K/V staging, out projection + residual.

#define NTOK 4096
#define CDIM 512
#define NBH  16

typedef __attribute__((ext_vector_type(16))) _Float16 v16h;
typedef __attribute__((ext_vector_type(8)))  _Float16 v8h;
typedef __attribute__((ext_vector_type(8)))  float    v8f;

// Types matching the async-load builtin's signature (v4i in AS1 / AS3).
typedef int v4i __attribute__((vector_size(16)));
typedef __attribute__((address_space(1))) v4i as1_v4i;
typedef __attribute__((address_space(3))) v4i as3_v4i;

#if defined(__has_builtin)
#if __has_builtin(__builtin_amdgcn_global_load_async_to_lds_b128)
#define HAVE_ASYNC_LDS 1
#endif
#endif
#ifndef HAVE_ASYNC_LDS
#define HAVE_ASYNC_LDS 0
#endif

// Pointer -> integer -> AS-qualified pointer (AS3 truncation = LDS offset).
#define AS1_V4I(p) ((as1_v4i*)(unsigned long long)(p))
#define AS3_V4I(p) ((as3_v4i*)(unsigned int)(unsigned long long)(p))

__device__ __forceinline__ v8f wmma16(v16h a, v16h b, v8f c) {
  return __builtin_amdgcn_wmma_f32_16x16x32_f16(
      false, a, false, b, (short)0, c, false, false);
}

// A-fragment (16x32 f16): lanes 0-15 hold K=[k0..k0+15] of row
// M=lane; lanes 16-31 hold K=[k0+16..k0+31]. Two 16B chunks.
__device__ __forceinline__ v16h load_a(const _Float16* p_lo, const _Float16* p_hi) {
  union { v16h v; v8h h[2]; } u;
  u.h[0] = *(const v8h*)p_lo;
  u.h[1] = *(const v8h*)p_hi;
  return u.v;
}
// B-fragment: 16 contiguous K-halves per lane, 16B-aligned source (LDS rows
// padded to 144B are only 16B aligned).
__device__ __forceinline__ v16h load_b16(const _Float16* p) {
  union { v16h v; v8h h[2]; } u;
  u.h[0] = *(const v8h*)p;
  u.h[1] = *(const v8h*)(p + 8);
  return u.v;
}

// ---------------------------------------------------------------- staging ---
__global__ void cvt_f32_f16(const float* __restrict__ in,
                            _Float16* __restrict__ out, int n) {
  int i = blockIdx.x * blockDim.x + threadIdx.x;
  if (i < n) out[i] = (_Float16)in[i];
}

// x[b,c,n] f32 -> xT[b,n,c] f16, LDS-tiled 64x64 (both sides coalesced)
__global__ __launch_bounds__(256) void transpose_x_f16(
    const float* __restrict__ x, _Float16* __restrict__ xT) {
  __shared__ _Float16 t[64][66];
  int bid = blockIdx.x;           // 2 * 8 ctiles * 64 ntiles = 1024
  int nt = bid & 63;
  int ct = (bid >> 6) & 7;
  int b  = bid >> 9;
  int tx = threadIdx.x & 63;
  int ty = threadIdx.x >> 6;      // 4 rows per pass
#pragma unroll
  for (int r = 0; r < 64; r += 4)
    t[r + ty][tx] =
        (_Float16)x[((size_t)b * CDIM + ct * 64 + r + ty) * NTOK + nt * 64 + tx];
  __syncthreads();
#pragma unroll
  for (int r = 0; r < 64; r += 4)
    xT[((size_t)b * NTOK + nt * 64 + r + ty) * CDIM + ct * 64 + tx] = t[tx][r + ty];
}

// --------------------------------------------------------------- QKV GEMM ---
// 32(o) x 64(n) per wave; routes epilogue to qT[b,h,n,d]*0.125, kT[b,h,m,d],
// v[b,h,d,m].
__global__ __launch_bounds__(256) void qkv_gemm(
    const _Float16* __restrict__ w16, const float* __restrict__ bqkv,
    const _Float16* __restrict__ xT,
    _Float16* __restrict__ qT, _Float16* __restrict__ kT,
    _Float16* __restrict__ vN) {
  int wid  = blockIdx.x * 8 + (threadIdx.x >> 5);
  int lane = threadIdx.x & 31;
  int half = lane >> 4, l15 = lane & 15;

  int nt  = wid & 63;
  int tmp = wid >> 6;
  int ot  = tmp % 48;            // 48 o-tiles of 32 (1536 rows)
  int b   = tmp / 48;
  int obase = ot * 32, nbase = nt * 64;

  const _Float16* arow0 = w16 + (size_t)(obase + l15) * CDIM;
  const _Float16* arow1 = arow0 + (size_t)16 * CDIM;

  v8f z = {};
  v8f acc[2][4] = {{z, z, z, z}, {z, z, z, z}};

  for (int c0 = 0; c0 < CDIM; c0 += 32) {
    if (c0 + 64 < CDIM) {
      __builtin_prefetch(arow0 + c0 + 64, 0, 3);
      __builtin_prefetch(arow1 + c0 + 64, 0, 3);
    }
    v16h a0 = load_a(arow0 + c0 + half * 16, arow0 + c0 + half * 16 + 8);
    v16h a1 = load_a(arow1 + c0 + half * 16, arow1 + c0 + half * 16 + 8);
#pragma unroll
    for (int s = 0; s < 4; ++s) {
      const _Float16* bp = xT + ((size_t)b * NTOK + nbase + s * 16 + l15) * CDIM
                              + c0 + half * 16;
      v16h bb = *(const v16h*)bp;
      acc[0][s] = wmma16(a0, bb, acc[0][s]);
      acc[1][s] = wmma16(a1, bb, acc[1][s]);
    }
  }

#pragma unroll
  for (int u = 0; u < 2; ++u)
#pragma unroll
    for (int s = 0; s < 4; ++s) {
      int n = nbase + s * 16 + l15;
#pragma unroll
      for (int j = 0; j < 8; ++j) {
        int o = obase + u * 16 + j + half * 8;
        float val = acc[u][s][j] + bqkv[o];
        int grp = o >> 9;
        int oo  = o & 511;
        int h = oo >> 6, d = oo & 63;
        size_t bh = (size_t)(b * 8 + h);
        if (grp == 0)      qT[(bh * NTOK + n) * 64 + d] = (_Float16)(val * 0.125f);
        else if (grp == 1) kT[(bh * NTOK + n) * 64 + d] = (_Float16)val;
        else               vN[(bh * 64 + d) * NTOK + n] = (_Float16)val;
      }
    }
}

// --------------------------------------------------------- flash attention ---
#define LROW 72   // padded LDS row stride (halves): 144B -> conflict-spread

// Cooperative stage of one 64-m K/V chunk into LDS (16 KB per block-chunk).
__device__ __forceinline__ void stage_kv(const _Float16* __restrict__ kbase,
                                         const _Float16* __restrict__ vbase,
                                         int mb, _Float16 (*ks)[LROW],
                                         _Float16 (*vs)[LROW], int tid) {
#pragma unroll
  for (int r = 0; r < 2; ++r) {
    int flat = tid + r * 256;         // 512 x 16B transfers per matrix
    int row  = flat >> 3;
    int col  = (flat & 7) * 8;
    const _Float16* gk = kbase + (size_t)(mb + row) * 64 + col;
    const _Float16* gv = vbase + (size_t)row * NTOK + mb + col;
#if HAVE_ASYNC_LDS
    __builtin_amdgcn_global_load_async_to_lds_b128(
        AS1_V4I(gk), AS3_V4I(&ks[row][col]), 0, 0);
    __builtin_amdgcn_global_load_async_to_lds_b128(
        AS1_V4I(gv), AS3_V4I(&vs[row][col]), 0, 0);
#else
    *(v8h*)&ks[row][col] = *(const v8h*)gk;
    *(v8h*)&vs[row][col] = *(const v8h*)gv;
#endif
  }
}

__device__ __forceinline__ void wait_async_done() {
#if HAVE_ASYNC_LDS
#if __has_builtin(__builtin_amdgcn_s_wait_asynccnt)
  __builtin_amdgcn_s_wait_asynccnt(0);
#else
  asm volatile("s_wait_asynccnt 0" ::: "memory");
#endif
#endif
}

// 8 waves/block share one (b,h); each wave owns a 16-row query tile; K/V
// double-buffered in LDS via async DMA; online softmax; P relaid D->A via a
// wave-private LDS tile guarded by s_wait_dscnt.
__global__ __launch_bounds__(256) void flash_attn(
    const _Float16* __restrict__ qT, const _Float16* __restrict__ kT,
    const _Float16* __restrict__ vN, _Float16* __restrict__ oT) {
  __shared__ __align__(32) _Float16 kst[2][64][LROW];
  __shared__ __align__(32) _Float16 vst[2][64][LROW];
  __shared__ __align__(32) _Float16 pbuf[8][16][LROW];

  int tid  = threadIdx.x;
  int w    = tid >> 5;
  int lane = tid & 31;
  int half = lane >> 4, l15 = lane & 15;

  int wid   = blockIdx.x * 8 + w;
  int ntile = wid & 255;             // all 8 waves of a block share bh
  int bh    = wid >> 8;
  int nbase = ntile * 16;
  int b = bh >> 3, h = bh & 7;

  const _Float16* kbase = kT + (size_t)bh * NTOK * 64;
  const _Float16* vbase = vN + (size_t)bh * 64 * NTOK;

  // Persistent Q A-fragments (K-dim = d: two 32-steps)
  const _Float16* qrow = qT + ((size_t)bh * NTOK + nbase + l15) * 64;
  v16h aq0 = load_a(qrow + half * 16,      qrow + half * 16 + 8);
  v16h aq1 = load_a(qrow + 32 + half * 16, qrow + 40 + half * 16);

  v8f z = {};
  v8f oacc[4] = {z, z, z, z};
  float rm[8], rl[8];
#pragma unroll
  for (int j = 0; j < 8; ++j) { rm[j] = -1e30f; rl[j] = 0.0f; }

  stage_kv(kbase, vbase, 0, kst[0], vst[0], tid);
  int buf = 0;

  for (int mb = 0; mb < NTOK; mb += 64) {
    wait_async_done();
    __syncthreads();                     // chunk (mb) resident in kst/vst[buf]
    if (mb + 64 < NTOK)                  // overlap next DMA with compute
      stage_kv(kbase, vbase, mb + 64, kst[buf ^ 1], vst[buf ^ 1], tid);

    _Float16 (*kb)[LROW] = kst[buf];
    _Float16 (*vb)[LROW] = vst[buf];

    // ---- S = Q*K^T (4 m-subtiles x 2 K-steps)
    v8f S[4];
#pragma unroll
    for (int s = 0; s < 4; ++s) {
      const _Float16* krow = &kb[s * 16 + l15][0];
      v8f t = z;
      t = wmma16(aq0, load_b16(krow + half * 16), t);
      t = wmma16(aq1, load_b16(krow + 32 + half * 16), t);
      S[s] = t;
    }

    // ---- online softmax (row = j + 8*half; butterfly over 16-lane half)
    float alpha[8];
#pragma unroll
    for (int j = 0; j < 8; ++j) {
      float mx = fmaxf(fmaxf(S[0][j], S[1][j]), fmaxf(S[2][j], S[3][j]));
#pragma unroll
      for (int off = 8; off >= 1; off >>= 1)
        mx = fmaxf(mx, __shfl_xor(mx, off));
      float nm = fmaxf(rm[j], mx);
      alpha[j] = __expf(rm[j] - nm);
      rm[j] = nm;
      float rs = 0.0f;
#pragma unroll
      for (int s = 0; s < 4; ++s) {
        float p = __expf(S[s][j] - nm);
        S[s][j] = p;
        rs += p;
      }
#pragma unroll
      for (int off = 8; off >= 1; off >>= 1)
        rs += __shfl_xor(rs, off);
      rl[j] = rl[j] * alpha[j] + rs;
    }
#pragma unroll
    for (int j = 0; j < 8; ++j) {
      oacc[0][j] *= alpha[j]; oacc[1][j] *= alpha[j];
      oacc[2][j] *= alpha[j]; oacc[3][j] *= alpha[j];
    }

    // ---- P: D-layout -> A-layout through wave-private LDS tile
#pragma unroll
    for (int s = 0; s < 4; ++s)
#pragma unroll
      for (int j = 0; j < 8; ++j)
        pbuf[w][j + half * 8][s * 16 + l15] = (_Float16)S[s][j];
    asm volatile("s_wait_dscnt 0" ::: "memory");
    const _Float16* prow = &pbuf[w][l15][0];
    v16h ap0 = load_a(prow + half * 16,      prow + half * 16 + 8);
    v16h ap1 = load_a(prow + 32 + half * 16, prow + 40 + half * 16);

    // ---- O += P * V^T (4 d-subtiles x 2 K-steps over m)
#pragma unroll
    for (int t = 0; t < 4; ++t) {
      const _Float16* vrow = &vb[t * 16 + l15][0];
      oacc[t] = wmma16(ap0, load_b16(vrow + half * 16), oacc[t]);
      oacc[t] = wmma16(ap1, load_b16(vrow + 32 + half * 16), oacc[t]);
    }
    buf ^= 1;
  }

  // ---- normalize, store oT[b, n, h*64+d] (B-layout for out projection)
#pragma unroll
  for (int j = 0; j < 8; ++j) {
    float inv = 1.0f / rl[j];
    int n = nbase + j + half * 8;
    size_t orow = ((size_t)b * NTOK + n) * CDIM + h * 64;
#pragma unroll
    for (int t = 0; t < 4; ++t)
      oT[orow + t * 16 + l15] = (_Float16)(oacc[t][j] * inv);
  }
}

// ---------------------------------------------------------- out projection ---
__global__ __launch_bounds__(256) void out_proj(
    const _Float16* __restrict__ wo16, const float* __restrict__ bout,
    const _Float16* __restrict__ oT, const float* __restrict__ x,
    const float* __restrict__ gamma, float* __restrict__ out) {
  int wid  = blockIdx.x * 8 + (threadIdx.x >> 5);
  int lane = threadIdx.x & 31;
  int half = lane >> 4, l15 = lane & 15;

  int nt  = wid & 63;
  int tmp = wid >> 6;
  int ot  = tmp & 15;            // 16 o-tiles of 32 (512 rows)
  int b   = tmp >> 4;
  int obase = ot * 32, nbase = nt * 64;
  float g = gamma[0];

  const _Float16* arow0 = wo16 + (size_t)(obase + l15) * CDIM;
  const _Float16* arow1 = arow0 + (size_t)16 * CDIM;

  v8f z = {};
  v8f acc[2][4] = {{z, z, z, z}, {z, z, z, z}};
  for (int c0 = 0; c0 < CDIM; c0 += 32) {
    if (c0 + 64 < CDIM) {
      __builtin_prefetch(arow0 + c0 + 64, 0, 3);
      __builtin_prefetch(arow1 + c0 + 64, 0, 3);
    }
    v16h a0 = load_a(arow0 + c0 + half * 16, arow0 + c0 + half * 16 + 8);
    v16h a1 = load_a(arow1 + c0 + half * 16, arow1 + c0 + half * 16 + 8);
#pragma unroll
    for (int s = 0; s < 4; ++s) {
      const _Float16* bp = oT + ((size_t)b * NTOK + nbase + s * 16 + l15) * CDIM
                              + c0 + half * 16;
      v16h bb = *(const v16h*)bp;
      acc[0][s] = wmma16(a0, bb, acc[0][s]);
      acc[1][s] = wmma16(a1, bb, acc[1][s]);
    }
  }

#pragma unroll
  for (int u = 0; u < 2; ++u)
#pragma unroll
    for (int s = 0; s < 4; ++s) {
      int n = nbase + s * 16 + l15;
#pragma unroll
      for (int j = 0; j < 8; ++j) {
        int o = obase + u * 16 + j + half * 8;
        size_t idx = ((size_t)b * CDIM + o) * NTOK + n;
        out[idx] = g * (acc[u][s][j] + bout[o]) + x[idx];
      }
    }
}

// ------------------------------------------------------------------ launch ---
extern "C" void kernel_launch(void* const* d_in, const int* in_sizes, int n_in,
                              void* d_out, int out_size, void* d_ws, size_t ws_size,
                              hipStream_t stream) {
  const float* x     = (const float*)d_in[0];
  const float* w_qkv = (const float*)d_in[1];
  const float* b_qkv = (const float*)d_in[2];
  const float* w_out = (const float*)d_in[3];
  const float* b_out = (const float*)d_in[4];
  const float* gamma = (const float*)d_in[5];
  float* out = (float*)d_out;

  _Float16* ws   = (_Float16*)d_ws;
  _Float16* xT   = ws;                                   // 2*4096*512
  _Float16* wq16 = xT   + (size_t)2 * NTOK * CDIM;       // 1536*512
  _Float16* wo16 = wq16 + (size_t)3 * CDIM * CDIM;       // 512*512
  _Float16* qTp  = wo16 + (size_t)CDIM * CDIM;           // 16*4096*64 each
  _Float16* kTp  = qTp  + (size_t)NBH * NTOK * 64;
  _Float16* vNp  = kTp  + (size_t)NBH * NTOK * 64;
  _Float16* oTp  = vNp  + (size_t)NBH * NTOK * 64;       // 2*4096*512

  cvt_f32_f16<<<(3 * CDIM * CDIM) / 256, 256, 0, stream>>>(w_qkv, wq16, 3 * CDIM * CDIM);
  cvt_f32_f16<<<(CDIM * CDIM) / 256, 256, 0, stream>>>(w_out, wo16, CDIM * CDIM);
  transpose_x_f16<<<1024, 256, 0, stream>>>(x, xT);

  // 2 * 48 o-tiles * 64 n-tiles = 6144 waves -> 768 blocks
  qkv_gemm<<<768, 256, 0, stream>>>(wq16, b_qkv, xT, qTp, kTp, vNp);
  // 16 (b,h) * 256 n-tiles = 4096 waves -> 512 blocks
  flash_attn<<<512, 256, 0, stream>>>(qTp, kTp, vNp, oTp);
  // 2 * 16 o-tiles * 64 n-tiles = 2048 waves -> 256 blocks
  out_proj<<<256, 256, 0, stream>>>(wo16, b_out, oTp, x, gamma, out);
}
